// RGCN_7997229105342
// MI455X (gfx1250) — matrix operand order repaired and checked by reading, not compiled, not run on hardware
//
#include <hip/hip_runtime.h>

#define NN 20000
#define NE 100000
#define DIM 768
#define NREL 4

#define LDSW 776              // 768 + 8 bf16 pad: 388 dwords/row -> bank offset 4/row
#define LDS_BYTES (64 * LDSW * 2)

typedef __bf16 bf16;
typedef bf16  v16bf __attribute__((ext_vector_type(16)));
typedef float v8f   __attribute__((ext_vector_type(8)));

struct Pack32 { uint4 lo, hi; };

static __device__ __forceinline__ v16bf load2_bf16x16(const bf16* p0, const bf16* p1) {
  Pack32 pk;
  pk.lo = *(const uint4*)p0;   // b128
  pk.hi = *(const uint4*)p1;   // b128
  return __builtin_bit_cast(v16bf, pk);
}

// ---------------------------------------------------------------------------
// GEMM: C[20000 x 768](f32) = A[20000 x 768](bf16) @ W, with Bt[n][k]=W[k][n] bf16.
// Block: 8 waves, 256 rows x 64 cols. B tile staged in LDS, shared by all waves.
// Wave: 32 rows x 64 cols = 8 f32 accumulators. Per 32-K step: 4 global b128 (A),
// 8 ds b128 (B), 8 WMMAs.
// A per lane-half holds a contiguous 16-K block of row m=lane%16; B per
// lane-half holds a contiguous 16-K block of column n=lane%16 (same K order).
// ---------------------------------------------------------------------------
__global__ __launch_bounds__(256) void rgcn_wmma_gemm(
    const bf16* __restrict__ A, const bf16* __restrict__ Bt,
    const float* __restrict__ bias, float* __restrict__ C, int useBias)
{
  extern __shared__ char smem_raw[];
  bf16* smem = (bf16*)smem_raw;

  const int NTG = DIM / 64;                  // 12 column groups
  const int blockM = blockIdx.x / NTG;
  const int tg     = blockIdx.x - blockM * NTG;

  // ---- stage B tile: rows [tg*64, tg*64+64) of Bt, all 768 K ----
  for (int i = threadIdx.x; i < 64 * (DIM / 8); i += 256) {
    int row = i / (DIM / 8);                 // 0..63
    int c8  = i - row * (DIM / 8);           // 0..95 (8-element chunks)
    uint4 v = *(const uint4*)(Bt + (size_t)(tg * 64 + row) * DIM + c8 * 8);
    *(uint4*)(smem + row * LDSW + c8 * 8) = v;   // ds_store_b128
  }
  __syncthreads();

  const int lane = threadIdx.x & 31;
  const int wave = threadIdx.x >> 5;
  const int lm = lane & 15;
  const int hi = lane >> 4;
  const int akh = hi * 16;                   // A contiguous 16-K block select
  const int bks = hi * 16;                   // B contiguous 16-K block select

  const int tmRaw0 = blockM * 16 + wave * 2; // row tiles (of 1250)
  const int tmRaw1 = tmRaw0 + 1;
  const int tm0 = tmRaw0 < 1250 ? tmRaw0 : 1249;   // clamp loads in tail block
  const int tm1 = tmRaw1 < 1250 ? tmRaw1 : 1249;

  const bf16* arow0 = A + (size_t)(tm0 * 16 + lm) * DIM;
  const bf16* arow1 = A + (size_t)(tm1 * 16 + lm) * DIM;
  const bf16* brow0 = smem + (0 * 16 + lm) * LDSW + bks;
  const bf16* brow1 = smem + (1 * 16 + lm) * LDSW + bks;
  const bf16* brow2 = smem + (2 * 16 + lm) * LDSW + bks;
  const bf16* brow3 = smem + (3 * 16 + lm) * LDSW + bks;

  v8f c00 = {}, c01 = {}, c02 = {}, c03 = {};
  v8f c10 = {}, c11 = {}, c12 = {}, c13 = {};

#pragma unroll 2
  for (int kk = 0; kk < DIM; kk += 32) {
    v16bf a0 = load2_bf16x16(arow0 + kk + akh, arow0 + kk + 8 + akh);
    v16bf a1 = load2_bf16x16(arow1 + kk + akh, arow1 + kk + 8 + akh);
    v16bf b0 = load2_bf16x16(brow0 + kk, brow0 + kk + 8);  // ds_load_b128 x2
    v16bf b1 = load2_bf16x16(brow1 + kk, brow1 + kk + 8);
    v16bf b2 = load2_bf16x16(brow2 + kk, brow2 + kk + 8);
    v16bf b3 = load2_bf16x16(brow3 + kk, brow3 + kk + 8);
    c00 = __builtin_amdgcn_wmma_f32_16x16x32_bf16(false, a0, false, b0, (short)0, c00, false, false);
    c01 = __builtin_amdgcn_wmma_f32_16x16x32_bf16(false, a0, false, b1, (short)0, c01, false, false);
    c02 = __builtin_amdgcn_wmma_f32_16x16x32_bf16(false, a0, false, b2, (short)0, c02, false, false);
    c03 = __builtin_amdgcn_wmma_f32_16x16x32_bf16(false, a0, false, b3, (short)0, c03, false, false);
    c10 = __builtin_amdgcn_wmma_f32_16x16x32_bf16(false, a1, false, b0, (short)0, c10, false, false);
    c11 = __builtin_amdgcn_wmma_f32_16x16x32_bf16(false, a1, false, b1, (short)0, c11, false, false);
    c12 = __builtin_amdgcn_wmma_f32_16x16x32_bf16(false, a1, false, b2, (short)0, c12, false, false);
    c13 = __builtin_amdgcn_wmma_f32_16x16x32_bf16(false, a1, false, b3, (short)0, c13, false, false);
  }

  // D layout: VGPR i, lanes 0-15 -> M=i, N=lane; lanes 16-31 -> M=i+8, N=lane-16.
  float bv0 = 0.f, bv1 = 0.f, bv2 = 0.f, bv3 = 0.f;
  if (useBias) {
    bv0 = bias[tg * 64 +  0 + lm];
    bv1 = bias[tg * 64 + 16 + lm];
    bv2 = bias[tg * 64 + 32 + lm];
    bv3 = bias[tg * 64 + 48 + lm];
  }
  if (tmRaw0 < 1250) {
    float* crow = C + (size_t)(tmRaw0 * 16 + hi * 8) * DIM + tg * 64 + lm;
#pragma unroll
    for (int i = 0; i < 8; ++i) {
      crow[(size_t)i * DIM +  0] = c00[i] + bv0;
      crow[(size_t)i * DIM + 16] = c01[i] + bv1;
      crow[(size_t)i * DIM + 32] = c02[i] + bv2;
      crow[(size_t)i * DIM + 48] = c03[i] + bv3;
    }
  }
  if (tmRaw1 < 1250) {
    float* crow = C + (size_t)(tmRaw1 * 16 + hi * 8) * DIM + tg * 64 + lm;
#pragma unroll
    for (int i = 0; i < 8; ++i) {
      crow[(size_t)i * DIM +  0] = c10[i] + bv0;
      crow[(size_t)i * DIM + 16] = c11[i] + bv1;
      crow[(size_t)i * DIM + 32] = c12[i] + bv2;
      crow[(size_t)i * DIM + 48] = c13[i] + bv3;
    }
  }
}

// ---------------------------------------------------------------------------
__global__ void zero_i32_kernel(int* __restrict__ p, int n) {
  int i = blockIdx.x * blockDim.x + threadIdx.x;
  if (i < n) p[i] = 0;
}

__global__ void count_deg_kernel(const int* __restrict__ dst,
                                 const int* __restrict__ et,
                                 int* __restrict__ deg) {
  int e = blockIdx.x * blockDim.x + threadIdx.x;
  if (e < NE) {
    __hip_atomic_fetch_add(&deg[et[e] * NN + dst[e]], 1,
                           __ATOMIC_RELAXED, __HIP_MEMORY_SCOPE_AGENT);
  }
}

// Wt[n][k] = (bf16) W[k][n]; one 768x768 matrix per launch.
__global__ void transpose_cvt_kernel(bf16* __restrict__ Wt,
                                     const float* __restrict__ W) {
  int idx = blockIdx.x * blockDim.x + threadIdx.x;
  if (idx < DIM * DIM) {
    int k = idx / DIM, n = idx - k * DIM;      // coalesced read over n
    Wt[(size_t)n * DIM + k] = (bf16)W[idx];
  }
}

__global__ void cvt_bf16_kernel(bf16* __restrict__ dst,
                                const float* __restrict__ src, int n) {
  int i = blockIdx.x * blockDim.x + threadIdx.x;
  if (i < n) dst[i] = (bf16)src[i];
}

__global__ void relu_kernel(float* __restrict__ x, int n) {
  int i = blockIdx.x * blockDim.x + threadIdx.x;
  if (i < n) x[i] = fmaxf(x[i], 0.0f);
}

// One block per edge: OUT[dst] += Hr[src] / deg_r[dst] for edges of relation r.
__global__ __launch_bounds__(256) void scatter_add_kernel(
    float* __restrict__ out, const float* __restrict__ H,
    const int* __restrict__ src, const int* __restrict__ dst,
    const int* __restrict__ et, const int* __restrict__ deg, int rel)
{
  int e = blockIdx.x;
  if (et[e] != rel) return;
  int s = src[e], d = dst[e];
  int dg = deg[rel * NN + d];
  float norm = 1.0f / (float)(dg > 1 ? dg : 1);
  const float* hs = H + (size_t)s * DIM;
  float* od = out + (size_t)d * DIM;
  for (int j = threadIdx.x; j < DIM; j += blockDim.x) {
    __hip_atomic_fetch_add(od + j, hs[j] * norm,
                           __ATOMIC_RELAXED, __HIP_MEMORY_SCOPE_AGENT);
  }
}

// ---------------------------------------------------------------------------
extern "C" void kernel_launch(void* const* d_in, const int* in_sizes, int n_in,
                              void* d_out, int out_size, void* d_ws, size_t ws_size,
                              hipStream_t stream) {
  (void)in_sizes; (void)n_in; (void)out_size; (void)ws_size;

  const float* entity = (const float*)d_in[0];
  const int*   eidx   = (const int*)d_in[1];
  const int*   etype  = (const int*)d_in[2];
  const int*   srcv   = eidx;
  const int*   dstv   = eidx + NE;
  const float* W[3]    = {(const float*)d_in[3], (const float*)d_in[6], (const float*)d_in[9]};
  const float* root[3] = {(const float*)d_in[4], (const float*)d_in[7], (const float*)d_in[10]};
  const float* bias[3] = {(const float*)d_in[5], (const float*)d_in[8], (const float*)d_in[11]};

  char* ws = (char*)d_ws;
  size_t off = 0;
  auto take = [&](size_t bytes) -> void* {
    void* p = ws + off;
    off = (off + bytes + 255) & ~(size_t)255;
    return p;
  };
  bf16*  Wt  = (bf16*)take((size_t)15 * DIM * DIM * sizeof(bf16));   // 15 transposed bf16 mats
  float* XA  = (float*)take((size_t)NN * DIM * sizeof(float));       // layer-1 output
  float* XB  = (float*)take((size_t)NN * DIM * sizeof(float));       // layer-2 output
  bf16*  Xbf = (bf16*)take((size_t)NN * DIM * sizeof(bf16));         // current x in bf16
  float* Hr  = (float*)take((size_t)NN * DIM * sizeof(float));       // per-relation transform
  int*   deg = (int*)take((size_t)NREL * NN * sizeof(int));          // per-relation in-degree

  const int T = 256;
  // Degrees (edge structure is layer-invariant).
  zero_i32_kernel<<<(NREL * NN + T - 1) / T, T, 0, stream>>>(deg, NREL * NN);
  count_deg_kernel<<<(NE + T - 1) / T, T, 0, stream>>>(dstv, etype, deg);

  // Transpose+convert all 15 weight matrices (layer-major: 4 relations + root).
  const int mg = (DIM * DIM + T - 1) / T;
  for (int L = 0; L < 3; ++L) {
    for (int s = 0; s < 5; ++s) {
      const float* srcmat = (s < 4) ? (W[L] + (size_t)s * DIM * DIM) : root[L];
      transpose_cvt_kernel<<<mg, T, 0, stream>>>(Wt + (size_t)(L * 5 + s) * DIM * DIM, srcmat);
    }
  }

  const int nElem = NN * DIM;
  const int mBlocks = (1250 + 15) / 16;            // 79 (wave = 2 row tiles, 8 waves)
  const int gemmBlocks = mBlocks * (DIM / 64);     // 79 * 12 = 948
  const float* Xcur = entity;
  float* outs[3] = {XA, XB, (float*)d_out};

  for (int L = 0; L < 3; ++L) {
    float* OUT = outs[L];
    cvt_bf16_kernel<<<(nElem + T - 1) / T, T, 0, stream>>>(Xbf, Xcur, nElem);
    // Self term: OUT = x @ root + bias
    rgcn_wmma_gemm<<<gemmBlocks, T, LDS_BYTES, stream>>>(
        Xbf, Wt + (size_t)(L * 5 + 4) * DIM * DIM, bias[L], OUT, 1);
    // Relation terms: mean-aggregated messages
    for (int r = 0; r < NREL; ++r) {
      rgcn_wmma_gemm<<<gemmBlocks, T, LDS_BYTES, stream>>>(
          Xbf, Wt + (size_t)(L * 5 + r) * DIM * DIM, nullptr, Hr, 0);
      scatter_add_kernel<<<NE, T, 0, stream>>>(OUT, Hr, srcv, dstv, etype, deg, r);
    }
    if (L < 2) relu_kernel<<<(nElem + T - 1) / T, T, 0, stream>>>(OUT, nElem);
    Xcur = OUT;
  }
}
